// DE_MSFT_50714973831308
// MI455X (gfx1250) — matrix operand order repaired and checked by reading, not compiled, not run on hardware
//
#include <hip/hip_runtime.h>
#include <hip/hip_bf16.h>

// ---------------------------------------------------------------------------
// Dual-encoder/decoder ViT forward for MI455X (gfx1250, wave32, WMMA).
// All GEMMs use v_wmma_f32_16x16x32_bf16. Weight-bandwidth-bound workload:
// ~610MB fp32 params, ~1 TFLOP of GEMMs -> stream weights as bf16, f32 accum.
// Interior GEMM tiles use 128-bit vectorized global->LDS staging (fast path);
// only M/N/K edge tiles (197-sized dims) take the guarded scalar path.
// ---------------------------------------------------------------------------

namespace {

constexpr int   kDim    = 768;
constexpr int   kHeads  = 12;
constexpr int   kDh     = 64;
constexpr int   kMlp    = 3072;
constexpr int   kTok    = 197;
constexpr int   kPatch  = 196;
constexpr int   kBatch  = 16;
constexpr int   kDepth  = 4;
constexpr int   kCls    = 1000;
constexpr int   kLdS    = 208;  // padded row stride for scores/probs (16B-aligned)
constexpr long long kRows  = (long long)kBatch * kTok;    // 3152
constexpr long long kPRows = (long long)kBatch * kPatch;  // 3136
constexpr float kAttScale = 0.03608439182435161f;         // 768^-0.5 (per reference)

typedef __attribute__((ext_vector_type(16))) __bf16 v16bf;
typedef __attribute__((ext_vector_type(8)))  float  v8f;

union FragB { v16bf v; unsigned int u[8]; };
union FragC { v8f   v; float        f[8]; };

__device__ __forceinline__ unsigned short f2bf(float f) {
  unsigned int u = __float_as_uint(f);
  u += 0x7fffu + ((u >> 16) & 1u);   // round-to-nearest-even
  return (unsigned short)(u >> 16);
}

// ---------------------------------------------------------------------------
// Batched bf16 WMMA GEMM:  C/D = scale * (A x B[^T]) + bias
//   A: [M,K] bf16 row-major (lda), batch offset zb*sAb + zh*sAh
//   B: !TRANSB -> [K,N] bf16 row-major (ldb); TRANSB -> [N,K] row-major (ldb)
//   C: optional f32 out, D: optional bf16 out. grid.z = Z = B*H batches.
// Block = 256 threads (8 waves) computes a 64x64 tile; each wave owns a
// 16(M) x 32(N) sub-tile = two 16x16x32 WMMA accumulators.
// Fast path requires lda/ldb % 8 == 0 and batch offsets % 8 == 0 (all callers
// satisfy this; scores/probs use padded stride 208).
// ---------------------------------------------------------------------------
template <int TRANSB>
__global__ __launch_bounds__(256) void k_gemm(
    const unsigned short* __restrict__ A, int lda, long long sAb, long long sAh,
    const unsigned short* __restrict__ B, int ldb, long long sBb, long long sBh,
    float* __restrict__ C, int ldc, long long sCb, long long sCh,
    unsigned short* __restrict__ D, int ldd, long long sDb, long long sDh,
    const float* __restrict__ bias, float scale,
    int M, int N, int K, int H) {
  __shared__ __align__(16) unsigned short As[64 * 32];  // [m][k]
  __shared__ __align__(16) unsigned short Bs[64 * 32];  // [n][k] (col-major store)

  const int z  = blockIdx.z;
  const int zb = z / H, zh = z % H;
  A += (long long)zb * sAb + (long long)zh * sAh;
  B += (long long)zb * sBb + (long long)zh * sBh;

  const int tid  = threadIdx.x;
  const int lane = tid & 31;
  const int wave = tid >> 5;
  const int half = lane >> 4;   // lanes 0-15 vs 16-31
  const int l16  = lane & 15;
  const int wm   = wave & 3;    // 4 row-stripes of 16
  const int wn   = wave >> 2;   // 2 col-stripes of 32
  const int m0   = blockIdx.y * 64;
  const int n0   = blockIdx.x * 64;

  const bool fullM = (m0 + 64 <= M);
  const bool fullN = (n0 + 64 <= N);

  FragC acc0, acc1;
#pragma unroll
  for (int i = 0; i < 8; ++i) { acc0.f[i] = 0.f; acc1.f[i] = 0.f; }

  for (int k0 = 0; k0 < K; k0 += 32) {
    const bool fullK = (k0 + 32 <= K);

    // ---- stage A tile 64x32 ----
    if (fullM && fullK) {
      int t8 = tid << 3;                     // 8 bf16 per thread, 2048 total
      int m = t8 >> 5, k = t8 & 31;
      *(uint4*)&As[t8] =
          *(const uint4*)(A + (long long)(m0 + m) * lda + (k0 + k));
    } else {
#pragma unroll
      for (int i = 0; i < 8; ++i) {
        int e = tid + 256 * i;
        int m = e >> 5, k = e & 31;
        int gm = m0 + m, gk = k0 + k;
        As[e] = (gm < M && gk < K) ? A[(long long)gm * lda + gk]
                                   : (unsigned short)0;
      }
    }

    // ---- stage B tile -> Bs[n][k] ----
    if (TRANSB) {
      if (fullN && fullK) {
        int t8 = tid << 3;
        int n = t8 >> 5, k = t8 & 31;
        *(uint4*)&Bs[t8] =
            *(const uint4*)(B + (long long)(n0 + n) * ldb + (k0 + k));
      } else {
#pragma unroll
        for (int i = 0; i < 8; ++i) {
          int e = tid + 256 * i;
          int n = e >> 5, k = e & 31;
          int gn = n0 + n, gk = k0 + k;
          Bs[n * 32 + k] = (gn < N && gk < K)
                               ? B[(long long)gn * ldb + gk]
                               : (unsigned short)0;
        }
      }
    } else {
      if (fullN && fullK) {
        int k = tid >> 3, n = (tid & 7) << 3;   // 32 k-rows x 64 n each 16B
        uint4 d = *(const uint4*)(B + (long long)(k0 + k) * ldb + (n0 + n));
        const unsigned short* e = (const unsigned short*)&d;
#pragma unroll
        for (int j = 0; j < 8; ++j) Bs[(n + j) * 32 + k] = e[j];
      } else {
#pragma unroll
        for (int i = 0; i < 8; ++i) {
          int e = tid + 256 * i;
          int k = e >> 6, n = e & 63;
          int gn = n0 + n, gk = k0 + k;
          Bs[n * 32 + k] = (gn < N && gk < K)
                               ? B[(long long)gk * ldb + gn]
                               : (unsigned short)0;
        }
      }
    }
    __syncthreads();

    // Fragment gather per CDNA5 16-bit A/B VGPR layouts (CDNA5 ISA §7.12.2)
    FragB fa, fb0, fb1;
#pragma unroll
    for (int v = 0; v < 8; ++v) {
      int ka = ((v < 4) ? 2 * v : 16 + 2 * (v - 4)) + 8 * half;  // A: K map
      fa.u[v] = *(const unsigned int*)&As[(wm * 16 + l16) * 32 + ka];
      int kb = ka;                                                // B: K map
      fb0.u[v] = *(const unsigned int*)&Bs[(wn * 32 + l16) * 32 + kb];
      fb1.u[v] = *(const unsigned int*)&Bs[(wn * 32 + 16 + l16) * 32 + kb];
    }
    acc0.v = __builtin_amdgcn_wmma_f32_16x16x32_bf16(false, fa.v, false, fb0.v,
                                                     (short)0, acc0.v, false, false);
    acc1.v = __builtin_amdgcn_wmma_f32_16x16x32_bf16(false, fa.v, false, fb1.v,
                                                     (short)0, acc1.v, false, false);
    __syncthreads();
  }

  if (C) C += (long long)zb * sCb + (long long)zh * sCh;
  if (D) D += (long long)zb * sDb + (long long)zh * sDh;
#pragma unroll
  for (int v = 0; v < 8; ++v) {            // C/D layout: M = v + 8*half
    int gm = m0 + wm * 16 + half * 8 + v;
    if (gm >= M) continue;
#pragma unroll
    for (int t = 0; t < 2; ++t) {
      int gn = n0 + wn * 32 + t * 16 + l16;
      if (gn >= N) continue;
      float val = (t ? acc1.f[v] : acc0.f[v]) * scale;
      if (bias) val += bias[gn];
      if (C) C[(long long)gm * ldc + gn] = val;
      if (D) D[(long long)gm * ldd + gn] = f2bf(val);
    }
  }
}

// ---------------------------------------------------------------------------
// Elementwise / reduction kernels
// ---------------------------------------------------------------------------
__global__ __launch_bounds__(256) void k_f32_to_bf16(
    const float* __restrict__ s, unsigned short* __restrict__ d, long long n) {
  long long i = (long long)blockIdx.x * blockDim.x + threadIdx.x;
  long long stride = (long long)gridDim.x * blockDim.x;
  for (; i < n; i += stride) d[i] = f2bf(s[i]);
}

// One wave per row of width kDim; f32 in -> bf16 out (feeds next GEMM).
__global__ __launch_bounds__(256) void k_layernorm(
    const float* __restrict__ x, const float* __restrict__ w,
    const float* __restrict__ b, unsigned short* __restrict__ y, int rows) {
  int row = blockIdx.x * 8 + (threadIdx.x >> 5);
  if (row >= rows) return;
  int lane = threadIdx.x & 31;
  const float* xr = x + (long long)row * kDim;
  float s = 0.f;
  for (int i = lane; i < kDim; i += 32) s += xr[i];
  for (int m = 16; m > 0; m >>= 1) s += __shfl_xor(s, m, 32);
  float mean = s * (1.0f / kDim);
  float var = 0.f;
  for (int i = lane; i < kDim; i += 32) { float d0 = xr[i] - mean; var += d0 * d0; }
  for (int m = 16; m > 0; m >>= 1) var += __shfl_xor(var, m, 32);
  float rstd = rsqrtf(var * (1.0f / kDim) + 1e-5f);
  unsigned short* yr = y + (long long)row * kDim;
  for (int i = lane; i < kDim; i += 32)
    yr[i] = f2bf((xr[i] - mean) * rstd * w[i] + b[i]);
}

// One wave per row; softmax over n (=197) with row stride ld; f32 -> bf16.
__global__ __launch_bounds__(256) void k_softmax(
    const float* __restrict__ x, unsigned short* __restrict__ y,
    long long rows, int n, int ld) {
  long long row = (long long)blockIdx.x * 8 + (threadIdx.x >> 5);
  if (row >= rows) return;
  int lane = threadIdx.x & 31;
  const float* xr = x + row * ld;
  float mx = -3.0e38f;
  for (int i = lane; i < n; i += 32) mx = fmaxf(mx, xr[i]);
  for (int m = 16; m > 0; m >>= 1) mx = fmaxf(mx, __shfl_xor(mx, m, 32));
  float s = 0.f;
  for (int i = lane; i < n; i += 32) s += __expf(xr[i] - mx);
  for (int m = 16; m > 0; m >>= 1) s += __shfl_xor(s, m, 32);
  float inv = 1.0f / s;
  unsigned short* yr = y + row * ld;
  for (int i = lane; i < n; i += 32) yr[i] = f2bf(__expf(xr[i] - mx) * inv);
}

__global__ __launch_bounds__(256) void k_gelu(
    const float* __restrict__ x, unsigned short* __restrict__ y, long long n) {
  long long i = (long long)blockIdx.x * blockDim.x + threadIdx.x;
  long long stride = (long long)gridDim.x * blockDim.x;
  for (; i < n; i += stride) {
    float v = x[i];
    y[i] = f2bf(0.5f * v * (1.0f + erff(v * 0.70710678118654752f)));
  }
}

__global__ __launch_bounds__(256) void k_add(
    float* __restrict__ dst, const float* __restrict__ src, long long n) {
  long long i = (long long)blockIdx.x * blockDim.x + threadIdx.x;
  long long stride = (long long)gridDim.x * blockDim.x;
  for (; i < n; i += stride) dst[i] += src[i];
}

__global__ __launch_bounds__(256) void k_add_out(
    float* __restrict__ dst, const float* __restrict__ a,
    const float* __restrict__ b, long long n) {
  long long i = (long long)blockIdx.x * blockDim.x + threadIdx.x;
  long long stride = (long long)gridDim.x * blockDim.x;
  for (; i < n; i += stride) dst[i] = a[i] + b[i];
}

// img[b,c,224,224] -> bf16 Xp[b*196 + patch, (p1*16+p2)*3 + c]
__global__ __launch_bounds__(256) void k_patchify(
    const float* __restrict__ img, unsigned short* __restrict__ xp) {
  long long n = kPRows * kDim;
  long long i = (long long)blockIdx.x * blockDim.x + threadIdx.x;
  long long stride = (long long)gridDim.x * blockDim.x;
  for (; i < n; i += stride) {
    int d = (int)(i % kDim);
    long long r = i / kDim;
    int patch = (int)(r % kPatch);
    int b = (int)(r / kPatch);
    int c = d % 3, pp = d / 3;
    int p2 = pp & 15, p1 = pp >> 4;
    int gx = patch % 14, gy = patch / 14;
    long long src = (((long long)b * 3 + c) * 224 + (gy * 16 + p1)) * 224 + (gx * 16 + p2);
    xp[i] = f2bf(img[src]);
  }
}

// x[b, t, :] = (t==0 ? cls : emb[b, t-1, :]) + pos[t, :]
__global__ __launch_bounds__(256) void k_assemble(
    const float* __restrict__ emb, const float* __restrict__ cls,
    const float* __restrict__ pos, float* __restrict__ x) {
  long long n = kRows * kDim;
  long long i = (long long)blockIdx.x * blockDim.x + threadIdx.x;
  long long stride = (long long)gridDim.x * blockDim.x;
  for (; i < n; i += stride) {
    int d = (int)(i % kDim);
    long long r = i / kDim;
    int t = (int)(r % kTok);
    long long b = r / kTok;
    float v = (t == 0) ? cls[d] : emb[(b * kPatch + (t - 1)) * kDim + d];
    x[i] = v + pos[(long long)t * kDim + d];
  }
}

__global__ __launch_bounds__(256) void k_cls_sum(
    const float* __restrict__ a, const float* __restrict__ b,
    float* __restrict__ o) {
  int i = blockIdx.x * 256 + threadIdx.x;
  if (i >= kBatch * kDim) return;
  int bb = i / kDim, d = i % kDim;
  long long off = (long long)bb * kTok * kDim + d;
  o[i] = a[off] + b[off];
}

}  // namespace

// ---------------------------------------------------------------------------
// Host orchestration
// ---------------------------------------------------------------------------
extern "C" void kernel_launch(void* const* d_in, const int* in_sizes, int n_in,
                              void* d_out, int out_size, void* d_ws, size_t ws_size,
                              hipStream_t stream) {
  (void)in_sizes; (void)n_in; (void)out_size; (void)ws_size;
  enum {
    IN_FEA_A = 0, IN_FEA_B, IN_POS, IN_CLS_A, IN_CLS_B,
    IN_PEAW, IN_PEAB, IN_PEBW, IN_PEBB,
    IN_EALNW, IN_EALNB, IN_EQKV, IN_EOW, IN_EOB,
    IN_EFLNW, IN_EFLNB, IN_EW1, IN_EB1, IN_EW2, IN_EB2,
    IN_DSLNW, IN_DSLNB, IN_DQKV, IN_DOW, IN_DOB,
    IN_DCLNW, IN_DCLNB, IN_DCQW, IN_DCKVW, IN_DCOW, IN_DCOB,
    IN_DFLNW, IN_DFLNB, IN_DW1, IN_DB1, IN_DW2, IN_DB2,
    IN_HLNW, IN_HLNB, IN_HW, IN_HB
  };
  auto F = [&](int i) { return (const float*)d_in[i]; };

  // bump allocator on workspace
  char* ws = (char*)d_ws;
  size_t off = 0;
  auto alloc = [&](size_t bytes) -> void* {
    void* p = ws + off;
    off += (bytes + 255) & ~(size_t)255;
    return p;
  };

  // ---- bf16 weights -------------------------------------------------------
  const long long nPe   = (long long)kDim * kDim;
  const long long nEqkv = (long long)kDepth * 2 * 2 * kDim * 3 * kDim;
  const long long nEow  = (long long)kDepth * 2 * 2 * kDim * kDim;
  const long long nEw1  = (long long)kDepth * 2 * kDim * kMlp;
  const long long nEw2  = (long long)kDepth * 2 * kMlp * kDim;
  const long long nDqkv = (long long)kDepth * 2 * kDim * 3 * kDim;
  const long long nDow  = (long long)kDepth * 2 * kDim * kDim;
  const long long nCqw  = (long long)kDepth * 2 * kDim * kDim;
  const long long nCkvw = (long long)kDepth * 2 * kDim * 2 * kDim;
  const long long nCow  = (long long)kDepth * 2 * kDim * kDim;
  const long long nHead = (long long)kDim * kCls;

  unsigned short* w_peA  = (unsigned short*)alloc(nPe * 2);
  unsigned short* w_peB  = (unsigned short*)alloc(nPe * 2);
  unsigned short* w_eqkv = (unsigned short*)alloc(nEqkv * 2);
  unsigned short* w_eow  = (unsigned short*)alloc(nEow * 2);
  unsigned short* w_ew1  = (unsigned short*)alloc(nEw1 * 2);
  unsigned short* w_ew2  = (unsigned short*)alloc(nEw2 * 2);
  unsigned short* w_dqkv = (unsigned short*)alloc(nDqkv * 2);
  unsigned short* w_dow  = (unsigned short*)alloc(nDow * 2);
  unsigned short* w_cqw  = (unsigned short*)alloc(nCqw * 2);
  unsigned short* w_ckvw = (unsigned short*)alloc(nCkvw * 2);
  unsigned short* w_cow  = (unsigned short*)alloc(nCow * 2);
  unsigned short* w_fw1  = (unsigned short*)alloc(nEw1 * 2);
  unsigned short* w_fw2  = (unsigned short*)alloc(nEw2 * 2);
  unsigned short* w_head = (unsigned short*)alloc(nHead * 2);

  auto cvt = [&](int idx, unsigned short* dst, long long n) {
    long long b = (n + 2047) / 2048; if (b > 4096) b = 4096;
    k_f32_to_bf16<<<dim3((unsigned)b), dim3(256), 0, stream>>>(F(idx), dst, n);
  };
  cvt(IN_PEAW, w_peA, nPe);     cvt(IN_PEBW, w_peB, nPe);
  cvt(IN_EQKV, w_eqkv, nEqkv);  cvt(IN_EOW, w_eow, nEow);
  cvt(IN_EW1,  w_ew1,  nEw1);   cvt(IN_EW2, w_ew2, nEw2);
  cvt(IN_DQKV, w_dqkv, nDqkv);  cvt(IN_DOW, w_dow, nDow);
  cvt(IN_DCQW, w_cqw,  nCqw);   cvt(IN_DCKVW, w_ckvw, nCkvw);
  cvt(IN_DCOW, w_cow,  nCow);
  cvt(IN_DW1,  w_fw1,  nEw1);   cvt(IN_DW2, w_fw2, nEw2);
  cvt(IN_HW,   w_head, nHead);

  // ---- activations --------------------------------------------------------
  float*          bufA      = (float*)alloc(kRows * kDim * 4);
  float*          bufB      = (float*)alloc(kRows * kDim * 4);
  float*          bufSpare  = (float*)alloc(kRows * kDim * 4);
  float*          proj      = (float*)alloc(kRows * kDim * 4);
  float*          ff1       = (float*)alloc(kRows * kMlp * 4);
  unsigned short* act_bf    = (unsigned short*)alloc(kRows * kMlp * 2);
  unsigned short* ln_q      = (unsigned short*)alloc(kRows * kDim * 2);
  unsigned short* ln_kv     = (unsigned short*)alloc(kRows * kDim * 2);
  unsigned short* qkv_bf    = (unsigned short*)alloc(kRows * 3 * kDim * 2);
  float*          scores    = (float*)alloc((long long)kBatch * kHeads * kTok * kLdS * 4);
  unsigned short* probs     = (unsigned short*)alloc((long long)kBatch * kHeads * kTok * kLdS * 2);
  unsigned short* concat_bf = (unsigned short*)alloc(kRows * kDim * 2);
  unsigned short* xp_bf     = (unsigned short*)alloc(kPRows * kDim * 2);
  float*          emb       = (float*)alloc(kPRows * kDim * 4);
  float*          clsE      = (float*)alloc(kBatch * kDim * 4);
  float*          clsD      = (float*)alloc(kBatch * kDim * 4);
  unsigned short* cls_ln    = (unsigned short*)alloc(kBatch * kDim * 2);
  unsigned short* q_bf      = qkv_bf;                       // [rows, 768]
  unsigned short* kv_bf     = qkv_bf + kRows * kDim;        // [rows, 1536]

  auto grid1d = [](long long n) {
    long long b = (n + 255) / 256; if (b > 32768) b = 32768;
    return dim3((unsigned)b, 1, 1);
  };

  auto gemm = [&](const unsigned short* A, int lda, long long sAb, long long sAh,
                  const unsigned short* B, int ldb, long long sBb, long long sBh,
                  int transB,
                  float* C, int ldc, long long sCb, long long sCh,
                  unsigned short* D, int ldd, long long sDb, long long sDh,
                  const float* bias, float scale, int M, int N, int K, int Z, int H) {
    dim3 g((N + 63) / 64, (M + 63) / 64, Z);
    if (transB)
      k_gemm<1><<<g, dim3(256), 0, stream>>>(A, lda, sAb, sAh, B, ldb, sBb, sBh,
                                             C, ldc, sCb, sCh, D, ldd, sDb, sDh,
                                             bias, scale, M, N, K, H);
    else
      k_gemm<0><<<g, dim3(256), 0, stream>>>(A, lda, sAb, sAh, B, ldb, sBb, sBh,
                                             C, ldc, sCb, sCh, D, ldd, sDb, sDh,
                                             bias, scale, M, N, K, H);
  };
  auto ln = [&](const float* x, const float* w, const float* b,
                unsigned short* y, int rows) {
    k_layernorm<<<dim3((rows + 7) / 8), dim3(256), 0, stream>>>(x, w, b, y, rows);
  };

  const long long sRowTok3 = (long long)kTok * 3 * kDim;  // qkv per-batch stride
  const long long sScoresH = (long long)kTok * kLdS;
  const long long sScoresB = (long long)kHeads * kTok * kLdS;

  // attention core: scores -> softmax -> A@V -> concat (bf16)
  auto attn_core = [&](const unsigned short* Q, int ldq, long long sQb,
                       const unsigned short* Kp, int ldk, long long sKb,
                       const unsigned short* Vp, int ldv, long long sVb) {
    gemm(Q, ldq, sQb, kDh, Kp, ldk, sKb, kDh, /*transB=*/1,
         scores, kLdS, sScoresB, sScoresH, nullptr, 0, 0, 0,
         nullptr, kAttScale, kTok, kTok, kDh, kBatch * kHeads, kHeads);
    long long srows = (long long)kBatch * kHeads * kTok;
    k_softmax<<<dim3((unsigned)((srows + 7) / 8)), dim3(256), 0, stream>>>(
        scores, probs, srows, kTok, kLdS);
    gemm(probs, kLdS, sScoresB, sScoresH, Vp, ldv, sVb, kDh, /*transB=*/0,
         nullptr, 0, 0, 0, concat_bf, kDim, (long long)kTok * kDim, kDh,
         nullptr, 1.f, kTok, kDh, kTok, kBatch * kHeads, kHeads);
  };

  // full pre-norm self-attention block, residual added in place
  auto self_attn = [&](float* x, const float* lnw, const float* lnb,
                       const unsigned short* wqkv, const unsigned short* wo,
                       const float* ob) {
    ln(x, lnw, lnb, ln_q, (int)kRows);
    gemm(ln_q, kDim, 0, 0, wqkv, 3 * kDim, 0, 0, 0,
         nullptr, 0, 0, 0, qkv_bf, 3 * kDim, 0, 0,
         nullptr, 1.f, (int)kRows, 3 * kDim, kDim, 1, 1);
    attn_core(qkv_bf,            3 * kDim, sRowTok3,
              qkv_bf + kDim,     3 * kDim, sRowTok3,
              qkv_bf + 2 * kDim, 3 * kDim, sRowTok3);
    gemm(concat_bf, kDim, 0, 0, wo, kDim, 0, 0, 0,
         proj, kDim, 0, 0, nullptr, 0, 0, 0,
         ob, 1.f, (int)kRows, kDim, kDim, 1, 1);
    k_add<<<grid1d(kRows * kDim), dim3(256), 0, stream>>>(x, proj, kRows * kDim);
  };

  // cross-attention: result (pre-residual) left in `proj`
  auto cross_attn = [&](const float* xq, const float* xkv,
                        const float* lnw, const float* lnb,
                        const unsigned short* wq, const unsigned short* wkv,
                        const unsigned short* wo, const float* ob) {
    ln(xq, lnw, lnb, ln_q, (int)kRows);
    ln(xkv, lnw, lnb, ln_kv, (int)kRows);
    gemm(ln_q, kDim, 0, 0, wq, kDim, 0, 0, 0,
         nullptr, 0, 0, 0, q_bf, kDim, 0, 0,
         nullptr, 1.f, (int)kRows, kDim, kDim, 1, 1);
    gemm(ln_kv, kDim, 0, 0, wkv, 2 * kDim, 0, 0, 0,
         nullptr, 0, 0, 0, kv_bf, 2 * kDim, 0, 0,
         nullptr, 1.f, (int)kRows, 2 * kDim, kDim, 1, 1);
    attn_core(q_bf,         kDim,     (long long)kTok * kDim,
              kv_bf,        2 * kDim, (long long)kTok * 2 * kDim,
              kv_bf + kDim, 2 * kDim, (long long)kTok * 2 * kDim);
    gemm(concat_bf, kDim, 0, 0, wo, kDim, 0, 0, 0,
         proj, kDim, 0, 0, nullptr, 0, 0, 0,
         ob, 1.f, (int)kRows, kDim, kDim, 1, 1);
  };

  auto ffn = [&](float* x, const float* lnw, const float* lnb,
                 const unsigned short* w1, const float* b1,
                 const unsigned short* w2, const float* b2) {
    ln(x, lnw, lnb, ln_q, (int)kRows);
    gemm(ln_q, kDim, 0, 0, w1, kMlp, 0, 0, 0,
         ff1, kMlp, 0, 0, nullptr, 0, 0, 0,
         b1, 1.f, (int)kRows, kMlp, kDim, 1, 1);
    k_gelu<<<grid1d(kRows * kMlp), dim3(256), 0, stream>>>(ff1, act_bf, kRows * kMlp);
    gemm(act_bf, kMlp, 0, 0, w2, kDim, 0, 0, 0,
         proj, kDim, 0, 0, nullptr, 0, 0, 0,
         b2, 1.f, (int)kRows, kDim, kMlp, 1, 1);
    k_add<<<grid1d(kRows * kDim), dim3(256), 0, stream>>>(x, proj, kRows * kDim);
  };

  // ---- patch embedding ----------------------------------------------------
  k_patchify<<<grid1d(kPRows * kDim), dim3(256), 0, stream>>>(F(IN_FEA_A), xp_bf);
  gemm(xp_bf, kDim, 0, 0, w_peA, kDim, 0, 0, 0,
       emb, kDim, 0, 0, nullptr, 0, 0, 0,
       F(IN_PEAB), 1.f, (int)kPRows, kDim, kDim, 1, 1);
  k_assemble<<<grid1d(kRows * kDim), dim3(256), 0, stream>>>(emb, F(IN_CLS_A), F(IN_POS), bufA);

  k_patchify<<<grid1d(kPRows * kDim), dim3(256), 0, stream>>>(F(IN_FEA_B), xp_bf);
  gemm(xp_bf, kDim, 0, 0, w_peB, kDim, 0, 0, 0,
       emb, kDim, 0, 0, nullptr, 0, 0, 0,
       F(IN_PEBB), 1.f, (int)kPRows, kDim, kDim, 1, 1);
  k_assemble<<<grid1d(kRows * kDim), dim3(256), 0, stream>>>(emb, F(IN_CLS_B), F(IN_POS), bufB);

  float* curA = bufA;
  float* curB = bufB;

  // ---- encoder ------------------------------------------------------------
  for (int i = 0; i < kDepth; ++i) {
    for (int s = 0; s < 2; ++s) {
      float* x = (s == 0) ? curA : curB;
      for (int j = 0; j < 2; ++j) {
        long long a = ((long long)(i * 2 + s) * 2 + j);
        self_attn(x, F(IN_EALNW) + a * kDim, F(IN_EALNB) + a * kDim,
                  w_eqkv + a * kDim * 3 * kDim, w_eow + a * kDim * kDim,
                  F(IN_EOB) + a * kDim);
      }
      long long f = (long long)(i * 2 + s);
      ffn(x, F(IN_EFLNW) + f * kDim, F(IN_EFLNB) + f * kDim,
          w_ew1 + f * kDim * kMlp, F(IN_EB1) + f * kMlp,
          w_ew2 + f * kMlp * kDim, F(IN_EB2) + f * kDim);
    }
  }
  k_cls_sum<<<dim3((kBatch * kDim + 255) / 256), dim3(256), 0, stream>>>(curA, curB, clsE);

  // ---- decoder ------------------------------------------------------------
  for (int i = 0; i < kDepth; ++i) {
    for (int s = 0; s < 2; ++s) {
      float* x = (s == 0) ? curA : curB;
      long long a = (long long)(i * 2 + s);
      self_attn(x, F(IN_DSLNW) + a * kDim, F(IN_DSLNB) + a * kDim,
                w_dqkv + a * kDim * 3 * kDim, w_dow + a * kDim * kDim,
                F(IN_DOB) + a * kDim);
    }
    // cross: A queries (pre-cross) B; B queries (pre-cross) A
    long long c0 = (long long)(i * 2 + 0), c1 = (long long)(i * 2 + 1);
    cross_attn(curA, curB, F(IN_DCLNW) + c0 * kDim, F(IN_DCLNB) + c0 * kDim,
               w_cqw + c0 * kDim * kDim, w_ckvw + c0 * kDim * 2 * kDim,
               w_cow + c0 * kDim * kDim, F(IN_DCOB) + c0 * kDim);
    k_add_out<<<grid1d(kRows * kDim), dim3(256), 0, stream>>>(bufSpare, curA, proj,
                                                              kRows * kDim);
    cross_attn(curB, curA, F(IN_DCLNW) + c1 * kDim, F(IN_DCLNB) + c1 * kDim,
               w_cqw + c1 * kDim * kDim, w_ckvw + c1 * kDim * 2 * kDim,
               w_cow + c1 * kDim * kDim, F(IN_DCOB) + c1 * kDim);
    k_add<<<grid1d(kRows * kDim), dim3(256), 0, stream>>>(curB, proj, kRows * kDim);
    float* tmp = curA; curA = bufSpare; bufSpare = tmp;   // commit xA2
    for (int s = 0; s < 2; ++s) {
      float* x = (s == 0) ? curA : curB;
      long long f = (long long)(i * 2 + s);
      ffn(x, F(IN_DFLNW) + f * kDim, F(IN_DFLNB) + f * kDim,
          w_fw1 + f * kDim * kMlp, F(IN_DB1) + f * kMlp,
          w_fw2 + f * kMlp * kDim, F(IN_DB2) + f * kDim);
    }
  }
  k_cls_sum<<<dim3((kBatch * kDim + 255) / 256), dim3(256), 0, stream>>>(curA, curB, clsD);

  // ---- heads --------------------------------------------------------------
  float* out = (float*)d_out;
  auto head = [&](const float* cls, float* o) {
    ln(cls, F(IN_HLNW), F(IN_HLNB), cls_ln, kBatch);
    gemm(cls_ln, kDim, 0, 0, w_head, kCls, 0, 0, 0,
         o, kCls, 0, 0, nullptr, 0, 0, 0,
         F(IN_HB), 1.f, kBatch, kCls, kDim, 1, 1);
  };
  head(clsE, out);
  head(clsD, out + (long long)kBatch * kCls);
}
